// DeepSetObstacles_56504589746465
// MI455X (gfx1250) — matrix operand order repaired and checked by reading, not compiled, not run on hardware
//
#include <hip/hip_runtime.h>

typedef _Float16 v16h __attribute__((ext_vector_type(16)));
typedef _Float16 v8h  __attribute__((ext_vector_type(8)));
typedef _Float16 v2h  __attribute__((ext_vector_type(2)));
typedef float    v8f  __attribute__((ext_vector_type(8)));

#define THREADS 256
#define STATES_PER_WG 8
#define ROWS 256            // STATES_PER_WG * 32 obstacles

// ---- WMMA fragment loaders (CDNA5 ISA 7.12.2 layouts) ----------------------
// A (16x32 f16) from row-major mat[m][k] (stride 64): lane L holds row
// M = L&15; VGPR0-3 halves = K in [hi*8..hi*8+7], VGPR4-7 = [16+hi*8..23+hi*8].
__device__ __forceinline__ v16h load_a_frag(const _Float16* base, int m, int k0, int lane) {
    int hi = (lane >> 4) << 3;                 // 0 or 8
    const _Float16* p = base + m * 64 + k0 + hi;
    union { v16h v; v8h h[2]; } u;
    u.h[0] = *(const v8h*)(p);                 // K = k0+hi    .. +7
    u.h[1] = *(const v8h*)(p + 16);            // K = k0+16+hi .. +7
    return u.v;
}
// B (32x16 f16): lane L holds column N = L&15; halves 0-7 = K in [k0+hi*8..+7],
// halves 8-15 = [k0+16+hi*8..+7] -> wants memory layout mat[n][k] (K contiguous).
__device__ __forceinline__ v16h load_b_frag(const _Float16* base, int n, int k0, int lane) {
    int khi = (lane >> 4) << 3;                // 0 or 8
    const _Float16* p = base + n * 64 + k0 + khi;
    union { v16h v; v8h h[2]; } u;
    u.h[0] = *(const v8h*)(p);
    u.h[1] = *(const v8h*)(p + 16);
    return u.v;
}

// One ROWSx64 layer:  dst = relu?(src @ W + b), computed as D = W^T x src^T.
// wt  = W^T row-major [out_ch][k] (stride 64)  -> A operand (hoisted, reused)
// src = activations row-major [row][k]         -> B operand
// dst = activations row-major [row][out_ch], one packed b128 store per tile.
// 4 independent accumulator chains per B-fragment -> dense XDL issue.
template <int NJ>
__device__ __forceinline__ void layer_wmma(const _Float16* __restrict__ src,
                                           const _Float16* __restrict__ wt,
                                           const float* __restrict__ bias,
                                           _Float16* __restrict__ dst,
                                           bool relu, int lane, int wave) {
    const int lm  = lane & 15;
    const int hi8 = (lane >> 4) << 3;
    v16h wa[4][2];
    v8f  bfrag[4];
#pragma unroll
    for (int ct = 0; ct < 4; ++ct) {
        wa[ct][0] = load_a_frag(wt, ct * 16 + lm, 0,  lane);
        wa[ct][1] = load_a_frag(wt, ct * 16 + lm, 32, lane);
        bfrag[ct] = *(const v8f*)(bias + ct * 16 + hi8);   // bias along M
    }
#pragma unroll
    for (int j = 0; j < NJ; ++j) {
        const int rt  = wave + j * 8;          // row-tile handled by this wave
        const int row = rt * 16 + lm;          // this lane's output row (N)
        v16h bb0 = load_b_frag(src, row, 0,  lane);
        v16h bb1 = load_b_frag(src, row, 32, lane);
        v8f c[4];
#pragma unroll
        for (int ct = 0; ct < 4; ++ct) c[ct] = bfrag[ct];
#pragma unroll
        for (int ct = 0; ct < 4; ++ct)
            c[ct] = __builtin_amdgcn_wmma_f32_16x16x32_f16(false, wa[ct][0], false, bb0, (short)0, c[ct], false, false);
#pragma unroll
        for (int ct = 0; ct < 4; ++ct)
            c[ct] = __builtin_amdgcn_wmma_f32_16x16x32_f16(false, wa[ct][1], false, bb1, (short)0, c[ct], false, false);
        _Float16* drow = dst + row * 64 + hi8;
#pragma unroll
        for (int ct = 0; ct < 4; ++ct) {
            v8h hv;
#pragma unroll
            for (int i = 0; i < 8; ++i) {
                float v = c[ct][i];
                if (relu) v = fmaxf(v, 0.f);
                hv[i] = (_Float16)v;
            }
            *(v8h*)(drow + ct * 16) = hv;      // 8 contiguous channels: 1x b128
        }
    }
}

extern "C" __global__ __launch_bounds__(THREADS)
void deepset_fused_kernel(const float* __restrict__ x, const float* __restrict__ vel,
                          const float* __restrict__ pW1, const float* __restrict__ pb1,
                          const float* __restrict__ pW2, const float* __restrict__ pb2,
                          const float* __restrict__ pW3, const float* __restrict__ pb3,
                          const float* __restrict__ rW1, const float* __restrict__ rb1,
                          const float* __restrict__ rW2, const float* __restrict__ rb2,
                          const float* __restrict__ rW3, const float* __restrict__ rb3,
                          float* __restrict__ out) {
    extern __shared__ __align__(16) char smem[];
    _Float16* actA  = (_Float16*)smem;           // 256*64
    _Float16* actB  = actA + ROWS * 64;          // 256*64
    _Float16* wbuf  = actB + ROWS * 64;          // 10240 halves, phased:
    _Float16* w2t   = wbuf;                      //   phi: W2^T | rho: rW1^T
    _Float16* w3t   = wbuf + 4096;               //   phi: W3^T | rho: rW2^T
    _Float16* r1t   = wbuf;
    _Float16* r2t   = wbuf + 4096;
    _Float16* r3t   = wbuf + 8192;               //   rho: rW3^T (32*64)
    _Float16* poolX = wbuf + 10240;              // 16*64 (rows 8..15 zeroed)
    _Float16* y1    = poolX + 1024;              // 16*64
    _Float16* y2    = y1 + 1024;                 // 16*64
    float* w1   = (float*)(y2 + 1024);           // 2*64 f32 (obstacle rows of W1)
    float* sb2  = w1 + 128;
    float* sb3  = sb2 + 64;
    float* sq1  = sb3 + 64;
    float* sq2  = sq1 + 64;
    float* sq3  = sq2 + 64;                      // 32
    float* velp = sq3 + 32;                      // 8*64: b1 + vel @ W1[2:4]

    const int t    = threadIdx.x;
    const int lane = t & 31;
    const int wave = t >> 5;
    const int wg   = blockIdx.x;
    const int stateBase = wg * STATES_PER_WG;
    const int lm  = lane & 15;
    const int hi8 = (lane >> 4) << 3;

    // ---- stage phi weights (transposed f16) + biases + velpart -------------
    for (int i = t; i < 4096; i += THREADS) {
        int n = i >> 6, k = i & 63;              // Wt[n*64+k] = W[k*dout+n]
        w2t[i] = (_Float16)pW2[k * 64 + n];
        w3t[i] = (_Float16)pW3[k * 64 + n];
    }
    if (t < 128) w1[t] = pW1[t];                 // rows k=0,1 (obstacle x,y)
    if (t < 64) { sb2[t] = pb2[t]; sb3[t] = pb3[t]; }
    {   // velp[s][n] = b1[n] + vx*W1[2][n] + vy*W1[3][n]  (global reads only)
        int i0 = 2 * t;                          // 512 entries, 2 per thread
        int sl = i0 >> 6;
        float vx = vel[(stateBase + sl) * 2 + 0];
        float vy = vel[(stateBase + sl) * 2 + 1];
#pragma unroll
        for (int j2 = 0; j2 < 2; ++j2) {
            int n = (i0 & 63) + j2;
            velp[sl * 64 + n] = pb1[n] + vx * pW1[128 + n] + vy * pW1[192 + n];
        }
    }
    __syncthreads();

    // ---- phi layer 1 (K=2 after velpart) VALU f32 -> actA (row-major) ------
    {
        int r  = t;                              // one row per thread
        int sl = r >> 5;
        int m  = r & 31;
        int st = stateBase + sl;
        float ox = x[st * 64 + 2 * m];
        float oy = x[st * 64 + 2 * m + 1];
        const float* vp = velp + sl * 64;
#pragma unroll
        for (int n = 0; n < 64; ++n) {
            float a = vp[n] + ox * w1[n] + oy * w1[64 + n];
            actA[r * 64 + n] = (_Float16)fmaxf(a, 0.f);
        }
    }
    __syncthreads();

    // ---- phi layer 2: relu(actA @ W2 + b2) -> actB -------------------------
    layer_wmma<2>(actA, w2t, sb2, actB, true, lane, wave);
    __syncthreads();

    // ---- phi layer 3 (linear): actB @ W3 + b3 -> actA ----------------------
    layer_wmma<2>(actB, w3t, sb3, actA, false, lane, wave);
    __syncthreads();

    // ---- pool over M=32 -> poolX(16x64, padded)  +  stage rho weights ------
    {
        int sl = t >> 5;                         // 0..7
        int n0 = (t & 31) * 2;                   // 2 cols per thread
        float s0 = 0.f, s1 = 0.f;
        const _Float16* p = actA + (sl * 32) * 64 + n0;
#pragma unroll
        for (int m = 0; m < 32; ++m) {
            v2h v = *(const v2h*)(p + m * 64);
            s0 += (float)v[0]; s1 += (float)v[1];
        }
        poolX[sl * 64 + n0 + 0] = (_Float16)s0;
        poolX[sl * 64 + n0 + 1] = (_Float16)s1;
        poolX[512 + 2 * t + 0] = (_Float16)0.f;  // zero pad rows 8..15
        poolX[512 + 2 * t + 1] = (_Float16)0.f;
    }
    for (int i = t; i < 4096; i += THREADS) {    // wbuf is dead here: reuse
        int n = i >> 6, k = i & 63;
        r1t[i] = (_Float16)rW1[k * 64 + n];
        r2t[i] = (_Float16)rW2[k * 64 + n];
    }
    for (int i = t; i < 2048; i += THREADS) {
        int n = i >> 6, k = i & 63;
        r3t[i] = (_Float16)rW3[k * 32 + n];
    }
    if (t < 64) { sq1[t] = rb1[t]; sq2[t] = rb2[t]; }
    if (t < 32) sq3[t] = rb3[t];
    __syncthreads();

    // ---- rho layer 1: relu(X @ rW1 + rb1) -> y1 ----------------------------
    if (wave < 4) {
        int ct = wave;
        v8f c = *(const v8f*)(sq1 + ct * 16 + hi8);
        v16h a0 = load_a_frag(r1t, ct * 16 + lm, 0,  lane);
        v16h b0 = load_b_frag(poolX, lm, 0,  lane);
        c = __builtin_amdgcn_wmma_f32_16x16x32_f16(false, a0, false, b0, (short)0, c, false, false);
        v16h a1 = load_a_frag(r1t, ct * 16 + lm, 32, lane);
        v16h b1 = load_b_frag(poolX, lm, 32, lane);
        c = __builtin_amdgcn_wmma_f32_16x16x32_f16(false, a1, false, b1, (short)0, c, false, false);
        v8h hv;
#pragma unroll
        for (int i = 0; i < 8; ++i) hv[i] = (_Float16)fmaxf(c[i], 0.f);
        *(v8h*)(y1 + lm * 64 + ct * 16 + hi8) = hv;
    }
    __syncthreads();
    // ---- rho layer 2: relu(y1 @ rW2 + rb2) -> y2 ---------------------------
    if (wave < 4) {
        int ct = wave;
        v8f c = *(const v8f*)(sq2 + ct * 16 + hi8);
        v16h a0 = load_a_frag(r2t, ct * 16 + lm, 0,  lane);
        v16h b0 = load_b_frag(y1, lm, 0,  lane);
        c = __builtin_amdgcn_wmma_f32_16x16x32_f16(false, a0, false, b0, (short)0, c, false, false);
        v16h a1 = load_a_frag(r2t, ct * 16 + lm, 32, lane);
        v16h b1 = load_b_frag(y1, lm, 32, lane);
        c = __builtin_amdgcn_wmma_f32_16x16x32_f16(false, a1, false, b1, (short)0, c, false, false);
        v8h hv;
#pragma unroll
        for (int i = 0; i < 8; ++i) hv[i] = (_Float16)fmaxf(c[i], 0.f);
        *(v8h*)(y2 + lm * 64 + ct * 16 + hi8) = hv;
    }
    __syncthreads();
    // ---- rho layer 3 (linear, dout=32): y2 @ rW3 + rb3 -> global -----------
    if (wave < 2) {
        int ct = wave;
        v8f c = *(const v8f*)(sq3 + ct * 16 + hi8);
        v16h a0 = load_a_frag(r3t, ct * 16 + lm, 0,  lane);
        v16h b0 = load_b_frag(y2, lm, 0,  lane);
        c = __builtin_amdgcn_wmma_f32_16x16x32_f16(false, a0, false, b0, (short)0, c, false, false);
        v16h a1 = load_a_frag(r3t, ct * 16 + lm, 32, lane);
        v16h b1 = load_b_frag(y2, lm, 32, lane);
        c = __builtin_amdgcn_wmma_f32_16x16x32_f16(false, a1, false, b1, (short)0, c, false, false);
        if (lm < STATES_PER_WG)                  // rows 8..15 are padding
            *(v8f*)(out + (stateBase + lm) * 32 + ct * 16 + hi8) = c;
    }
}

extern "C" void kernel_launch(void* const* d_in, const int* in_sizes, int n_in,
                              void* d_out, int out_size, void* d_ws, size_t ws_size,
                              hipStream_t stream) {
    (void)n_in; (void)d_ws; (void)ws_size; (void)out_size;
    const float* x   = (const float*)d_in[0];
    const float* vel = (const float*)d_in[1];
    const float* pW1 = (const float*)d_in[2];
    const float* pb1 = (const float*)d_in[3];
    const float* pW2 = (const float*)d_in[4];
    const float* pb2 = (const float*)d_in[5];
    const float* pW3 = (const float*)d_in[6];
    const float* pb3 = (const float*)d_in[7];
    const float* rW1 = (const float*)d_in[8];
    const float* rb1 = (const float*)d_in[9];
    const float* rW2 = (const float*)d_in[10];
    const float* rb2 = (const float*)d_in[11];
    const float* rW3 = (const float*)d_in[12];
    const float* rb3 = (const float*)d_in[13];
    float* out = (float*)d_out;

    const int Bn   = in_sizes[0] / 64;           // 65536 states
    const int grid = Bn / STATES_PER_WG;         // 8192 workgroups

    // LDS: halves (2*16384 + 10240 + 3*1024) = 46080 -> 92160 B
    //      floats (128 + 64*4 + 32 + 512)    =   928 ->  3712 B
    // total 95,872 B -> 3 workgroups per 320 KB WGP
    const size_t smem = (size_t)((ROWS * 64 * 2 + 10240 + 3 * 1024) * 2
                                 + (128 + 64 * 4 + 32 + 512) * 4);

    deepset_fused_kernel<<<grid, THREADS, smem, stream>>>(
        x, vel, pW1, pb1, pW2, pb2, pW3, pb3,
        rW1, rb1, rW2, rb2, rW3, rb3, out);
}
